// PrototypicalLoss_43422119362706
// MI455X (gfx1250) — matrix operand order repaired and checked by reading, not compiled, not run on hardware
//
#include <hip/hip_runtime.h>

// Episode constants (match the reference)
#define NCLS   2048
#define NSUP   5
#define NQRY   25
#define DIM    64
#define NSEC   32
#define PERC   30                    // NSUP + NQRY
#define NQTOT  51200                 // NCLS * NQRY
#define NTILE_C 128                  // NCLS / 16
#define NTILE_Q 3200                 // NQTOT / 16
#define SMEM_MAIN (NCLS*DIM*2 + NCLS*4 + NCLS*4)   // 278528 B < 320 KB WGP LDS

#define LOG2E 1.44269504088896f
#define LN2   0.69314718055995f

typedef __attribute__((ext_vector_type(16))) __bf16 v16bf;
typedef __attribute__((ext_vector_type(8)))  float  v8f;

struct __attribute__((aligned(16))) U4 { unsigned int x, y, z, w; };
union AB32 { U4 u[2]; v16bf f; };     // 32-byte WMMA A/B fragment (16 bf16)

__device__ __forceinline__ unsigned short f2bf(float v) {
  unsigned int u = __builtin_bit_cast(unsigned int, v);
  u += 0x7FFFu + ((u >> 16) & 1u);    // round-to-nearest-even
  return (unsigned short)(u >> 16);
}
__device__ __forceinline__ float bf2f(unsigned short h) {
  unsigned int u = ((unsigned int)h) << 16;
  return __builtin_bit_cast(float, u);
}

// ---- k0: zero sector accumulators and loss/acc accumulators -----------------
__global__ void kInit(float* secP, float* secCnt, float* accum) {
  int i = blockIdx.x * blockDim.x + threadIdx.x;
  if (i < NSEC * DIM) secP[i] = 0.f;
  if (i < NSEC)       secCnt[i] = 0.f;
  if (i < 2)          accum[i] = 0.f;
}

// ---- k1: class prototypes (mean of NSUP supports), bf16 copy, |p|^2, sector sums
__global__ void kProto(const float* __restrict__ x, const int* __restrict__ csec,
                       unsigned short* __restrict__ protoB, float* __restrict__ pnorm,
                       float* secP, float* secCnt) {
  int c = blockIdx.x, d = threadIdx.x;   // 64 threads
  const float* base = x + (size_t)c * PERC * DIM + d;
  float s = 0.f;
  #pragma unroll
  for (int j = 0; j < NSUP; ++j) s += base[j * DIM];
  float p = s * (1.0f / NSUP);
  protoB[c * DIM + d] = f2bf(p);
  int sec = csec[c];
  atomicAdd(&secP[sec * DIM + d], p);
  __shared__ float red[DIM];
  red[d] = p * p;
  __syncthreads();
  for (int off = DIM / 2; off > 0; off >>= 1) {
    if (d < off) red[d] += red[d + off];
    __syncthreads();
  }
  if (d == 0) { pnorm[c] = red[0]; atomicAdd(&secCnt[sec], 1.f); }
}

// ---- k2: sector prototype means --------------------------------------------
__global__ void kSecDiv(float* secP, const float* __restrict__ secCnt) {
  int i = blockIdx.x * blockDim.x + threadIdx.x;
  if (i < NSEC * DIM) secP[i] /= secCnt[i / DIM];
}

// ---- k3: per-query bf16 copy, |q|^2, nearest sector, true class, d(q, proto_true)
__global__ void kQprep(const float* __restrict__ x, const int* __restrict__ target,
                       const float* __restrict__ secP,
                       const unsigned short* __restrict__ protoB,
                       const float* __restrict__ pnorm,
                       unsigned short* __restrict__ qB, float* __restrict__ qnorm,
                       int* __restrict__ qsec, int* __restrict__ qtrue,
                       float* __restrict__ dtrue) {
  __shared__ float sSec[NSEC * DIM];                 // 8 KB
  int tid = threadIdx.x;
  for (int i = tid; i < NSEC * DIM; i += 256) sSec[i] = secP[i];
  __syncthreads();
  int wv = tid >> 5, lane = tid & 31;                // one wave32 per query
  int q = blockIdx.x * 8 + wv;
  int cls = q / NQRY, j = q % NQRY;
  int row = cls * PERC + NSUP + j;
  const float* qr = x + (size_t)row * DIM;
  float v0 = qr[lane], v1 = qr[lane + 32];
  unsigned short h0 = f2bf(v0), h1 = f2bf(v1);
  qB[(size_t)q * DIM + lane]      = h0;
  qB[(size_t)q * DIM + lane + 32] = h1;
  float qn = v0 * v0 + v1 * v1;
  // dot(q_bf16, proto_true_bf16): same quantization as the WMMA path
  float p0 = bf2f(protoB[cls * DIM + lane]), p1 = bf2f(protoB[cls * DIM + lane + 32]);
  float dt = bf2f(h0) * p0 + bf2f(h1) * p1;
  for (int mk = 16; mk; mk >>= 1) {
    qn += __shfl_xor(qn, mk, 32);
    dt += __shfl_xor(dt, mk, 32);
  }
  float d2 = 0.f;                                     // lane == sector id
  #pragma unroll 8
  for (int d = 0; d < DIM; ++d) { float df = qr[d] - sSec[lane * DIM + d]; d2 += df * df; }
  float bv = d2; int bi = lane;
  for (int mk = 16; mk; mk >>= 1) {
    float ov = __shfl_xor(bv, mk, 32); int oi = __shfl_xor(bi, mk, 32);
    bool take = (ov < bv) | ((ov == bv) & (oi < bi));
    bv = take ? ov : bv; bi = take ? oi : bi;
  }
  if (lane == 0) {
    qnorm[q] = qn; qsec[q] = bi; qtrue[q] = target[row];
    dtrue[q] = __builtin_amdgcn_sqrtf(fmaxf(qn + pnorm[cls] - 2.0f * dt, 0.f));
  }
}

// ---- k4: fused WMMA distance / online softmax / hierarchical argmin --------
__global__ __launch_bounds__(256) void kMain(
    const unsigned short* __restrict__ qB, const float* __restrict__ qnorm,
    const int* __restrict__ qsec, const int* __restrict__ qtrue,
    const float* __restrict__ dtrue,
    const unsigned short* __restrict__ protoB, const float* __restrict__ pnorm,
    const int* __restrict__ csec, float* accum) {
  extern __shared__ char smem[];
  unsigned short* sP = (unsigned short*)smem;                 // 256 KB bf16 protos
  float* sN = (float*)(smem + NCLS * DIM * 2);                // 8 KB |p|^2
  int*   sS = (int*)(smem + NCLS * DIM * 2 + NCLS * 4);       // 8 KB sector ids

  int tid = threadIdx.x;
  {
    // Async direct global->LDS staging of all prototypes (ASYNCcnt path).
    // LDS byte offset = low 32 bits of the generic shared-aperture address.
    unsigned lds0 = (unsigned)(size_t)(void*)sP;
    unsigned long long g0 = (unsigned long long)(const void*)protoB;
    for (int i = tid; i < NCLS * DIM * 2 / 16; i += 256) {
      unsigned lofs = lds0 + (unsigned)i * 16u;
      unsigned long long ga = g0 + (unsigned long long)i * 16ull;
      asm volatile("global_load_async_to_lds_b128 %0, %1, off"
                   :: "v"(lofs), "v"(ga) : "memory");
    }
    for (int i = tid; i < NCLS; i += 256) { sN[i] = pnorm[i]; sS[i] = csec[i]; }
    asm volatile("s_wait_asynccnt 0x0" ::: "memory");
  }
  __syncthreads();

  const int wv = tid >> 5, lane = tid & 31;
  const int hv = lane >> 4, ln = lane & 15;    // hv: half (lanes 0-15 vs 16-31)
  const int gw = blockIdx.x * 8 + wv;
  const int nw = gridDim.x * 8;

  for (int t = gw; t < NTILE_Q; t += nw) {     // one 16-query tile per wave
    const int q0 = t * 16;
    // A fragments (16x32 bf16 layout: lane ln holds row M=ln; hv selects K phase)
    const unsigned short* arow = qB + (size_t)(q0 + ln) * DIM;
    AB32 A0, A1;
    A0.u[0] = *(const U4*)(arow + hv * 8);
    A0.u[1] = *(const U4*)(arow + 16 + hv * 8);
    A1.u[0] = *(const U4*)(arow + 32 + hv * 8);
    A1.u[1] = *(const U4*)(arow + 48 + hv * 8);

    float qn[8], ssum[8];
    unsigned key[8];                           // (dist_bits & ~0x7FF) | class_id
    int qt_[8], qs_[8];
    #pragma unroll
    for (int r = 0; r < 8; ++r) {              // lane owns rows M = hv*8 + r
      int gq = q0 + hv * 8 + r;
      qn[r] = qnorm[gq]; qt_[r] = qtrue[gq]; qs_[r] = qsec[gq];
      ssum[r] = 0.f; key[r] = 0xFFFFFFFFu;
    }

    for (int nt = 0; nt < NTILE_C; ++nt) {     // sweep all 2048 classes
      const int n = nt * 16 + ln;              // this lane's column (class id)
      const unsigned short* brow = sP + n * DIM;
      AB32 B0, B1;                             // B 32x16: lane ln = N, hv = K phase
      B0.u[0] = *(const U4*)(brow + hv * 8);
      B0.u[1] = *(const U4*)(brow + 16 + hv * 8);
      B1.u[0] = *(const U4*)(brow + 32 + hv * 8);
      B1.u[1] = *(const U4*)(brow + 48 + hv * 8);
      const float pn = sN[n];
      const int   sc = sS[n];
      v8f acc = {};
      acc = __builtin_amdgcn_wmma_f32_16x16x32_bf16(false, A0.f, false, B0.f,
                                                    (short)0, acc, false, false);
      acc = __builtin_amdgcn_wmma_f32_16x16x32_bf16(false, A1.f, false, B1.f,
                                                    (short)0, acc, false, false);
      #pragma unroll
      for (int r = 0; r < 8; ++r) {
        float d2 = qn[r] + pn - 2.0f * acc[r];
        // raw-rate v_sqrt_f32 / v_exp_f32 (no libm fixup sequences)
        float dd = __builtin_amdgcn_sqrtf(fmaxf(d2, 0.f));
        ssum[r] += __builtin_amdgcn_exp2f(dd * -LOG2E);   // exp(-d)
        // argmin via monotone bit-packing: dd >= 0 so IEEE bits preserve order;
        // class id in the 11 LSBs gives lowest-index tie-break for free.
        unsigned kc = (__builtin_bit_cast(unsigned, dd) & 0xFFFFF800u) | (unsigned)n;
        kc = (sc == qs_[r]) ? kc : 0xFFFFFFFFu;
        key[r] = key[r] < kc ? key[r] : kc;
      }
    }

    // reduce across the 16 lanes of this half (class columns n % 16 == ln)
    float lsum = 0.f, asum = 0.f;
    #pragma unroll
    for (int r = 0; r < 8; ++r) {
      float s = ssum[r];
      unsigned k = key[r];
      #pragma unroll
      for (int mk = 1; mk <= 8; mk <<= 1) {
        s += __shfl_xor(s, mk, 32);
        unsigned ok = (unsigned)__shfl_xor((int)k, mk, 32);
        k = k < ok ? k : ok;
      }
      // -logp[true] = d_true + log(sum exp(-d));  log(s) = log2(s) * ln2
      lsum += dtrue[q0 + hv * 8 + r] + __builtin_amdgcn_logf(s) * LN2;
      asum += ((int)(k & 0x7FFu) == qt_[r]) ? 1.f : 0.f;
    }
    if (ln == 0) {                             // one writer per half-wave (8 rows each)
      atomicAdd(&accum[0], lsum);
      atomicAdd(&accum[1], asum);
    }
  }
}

// ---- k5: finalize ----------------------------------------------------------
__global__ void kFinal(const float* __restrict__ accum, float* __restrict__ out) {
  if (threadIdx.x == 0) {
    out[0] = accum[0] * (1.0f / NQTOT);
    out[1] = accum[1] * (1.0f / NQTOT);
  }
}

extern "C" void kernel_launch(void* const* d_in, const int* in_sizes, int n_in,
                              void* d_out, int out_size, void* d_ws, size_t ws_size,
                              hipStream_t stream) {
  (void)in_sizes; (void)n_in; (void)out_size; (void)ws_size;
  const float* x      = (const float*)d_in[0];
  const int*   target = (const int*)d_in[1];
  const int*   csec   = (const int*)d_in[2];
  float* out = (float*)d_out;
  char*  ws  = (char*)d_ws;

  size_t off = 0;
  auto alc = [&](size_t n) { size_t c = off; off += (n + 255) & ~(size_t)255; return c; };
  unsigned short* protoB = (unsigned short*)(ws + alc((size_t)NCLS * DIM * 2));
  float* pnorm  = (float*)(ws + alc((size_t)NCLS * 4));
  float* secP   = (float*)(ws + alc((size_t)NSEC * DIM * 4));
  float* secCnt = (float*)(ws + alc((size_t)NSEC * 4));
  unsigned short* qB = (unsigned short*)(ws + alc((size_t)NQTOT * DIM * 2));
  float* qnorm  = (float*)(ws + alc((size_t)NQTOT * 4));
  int*   qsec   = (int*)(ws + alc((size_t)NQTOT * 4));
  int*   qtrue  = (int*)(ws + alc((size_t)NQTOT * 4));
  float* dtrue  = (float*)(ws + alc((size_t)NQTOT * 4));
  float* accum  = (float*)(ws + alc(2 * 4));

  // allow >64KB dynamic LDS for the main kernel
  (void)hipFuncSetAttribute((const void*)kMain,
                            hipFuncAttributeMaxDynamicSharedMemorySize, SMEM_MAIN);

  kInit  <<<dim3((NSEC * DIM + 255) / 256), dim3(256), 0, stream>>>(secP, secCnt, accum);
  kProto <<<dim3(NCLS), dim3(DIM), 0, stream>>>(x, csec, protoB, pnorm, secP, secCnt);
  kSecDiv<<<dim3((NSEC * DIM + 255) / 256), dim3(256), 0, stream>>>(secP, secCnt);
  kQprep <<<dim3(NQTOT / 8), dim3(256), 0, stream>>>(x, target, secP, protoB, pnorm,
                                                     qB, qnorm, qsec, qtrue, dtrue);
  kMain  <<<dim3(128), dim3(256), SMEM_MAIN, stream>>>(qB, qnorm, qsec, qtrue, dtrue,
                                                       protoB, pnorm, csec, accum);
  kFinal <<<dim3(1), dim3(32), 0, stream>>>(accum, out);
}
